// Swin_4947802325757
// MI455X (gfx1250) — matrix operand order repaired and checked by reading, not compiled, not run on hardware
//
#include <hip/hip_runtime.h>

typedef _Float16 half_t;
typedef __attribute__((ext_vector_type(16))) _Float16 v16h;
typedef __attribute__((ext_vector_type(8)))  float    v8f;
typedef __attribute__((ext_vector_type(4)))  float    vf4;

#define NH_   8
#define HD_   40
#define C_    320
#define B_    2
#define NW_   1024          // windows per image (32x32)
#define BW_   (B_*NW_)      // 2048 windows total
#define N_    64            // tokens per window
#define M_TOT (BW_*N_)      // 131072 rows
#define SH_   4

// ---------------- WMMA helpers (CDNA5 gfx1250, wave32) ----------------

__device__ __forceinline__ v8f wmma16(v16h a, v16h b, v8f c) {
  return __builtin_amdgcn_wmma_f32_16x16x32_f16(false, a, false, b, (short)0, c,
                                                false, false);
}

// A-fragment (16xK rows, row-major, stride in elements). lane r = row M.
// K pattern: {k0+hf*8 .. +7} and {k0+16+hf*8 .. +7}
__device__ __forceinline__ v16h load_a16(const half_t* base, int stride, int k0,
                                         int r, int hf) {
  const half_t* p = base + (size_t)r * stride + k0 + hf * 8;
  v16h out;
  ((vf4*)&out)[0] = *(const vf4*)(p);
  ((vf4*)&out)[1] = *(const vf4*)(p + 16);
  return out;
}

// B-fragment from B^T stored row-major (N rows of K). lane r = column N.
// K pattern: {k0+hf*8 .. +7} and {k0+16+hf*8 .. +7}
__device__ __forceinline__ v16h load_b16(const half_t* baseT, int stride, int k0,
                                         int r, int hf) {
  const half_t* p = baseT + (size_t)r * stride + k0 + hf * 8;
  v16h out;
  ((vf4*)&out)[0] = *(const vf4*)(p);
  ((vf4*)&out)[1] = *(const vf4*)(p + 16);
  return out;
}

// ---------------- weight fp32 -> fp16 ----------------
__global__ __launch_bounds__(256) void k_cvt_h(const float* __restrict__ s,
                                               half_t* __restrict__ d, int n) {
  int i = blockIdx.x * 256 + threadIdx.x;
  if (i < n) d[i] = (half_t)s[i];
}

// ---------------- LN1 + roll(-4,-4) + window partition -> xw (f16) -------
__global__ __launch_bounds__(256) void k_ln1(const float* __restrict__ x,
                                             const float* __restrict__ g,
                                             const float* __restrict__ bta,
                                             half_t* __restrict__ xw) {
  int wave = blockIdx.x * 8 + (threadIdx.x >> 5);
  int lane = threadIdx.x & 31;
  int b = wave >> 16;
  int pix = wave & 65535;
  int y = pix >> 8, xc = pix & 255;
  const float* row = x + (size_t)wave * C_;
  float v[10];
  float s = 0.f;
#pragma unroll
  for (int i = 0; i < 10; i++) { v[i] = row[i * 32 + lane]; s += v[i]; }
#pragma unroll
  for (int d = 16; d >= 1; d >>= 1) s += __shfl_xor(s, d, 32);
  float mu = s * (1.0f / C_);
  float vs = 0.f;
#pragma unroll
  for (int i = 0; i < 10; i++) { float t = v[i] - mu; vs += t * t; }
#pragma unroll
  for (int d = 16; d >= 1; d >>= 1) vs += __shfl_xor(vs, d, 32);
  float rinv = rsqrtf(vs * (1.0f / C_) + 1e-5f);

  int yp = (y - SH_) & 255, xp = (xc - SH_) & 255;
  int widx = b * NW_ + (yp >> 3) * 32 + (xp >> 3);
  int n = (yp & 7) * 8 + (xp & 7);
  half_t* dst = xw + (size_t)(widx * 64 + n) * C_;
#pragma unroll
  for (int i = 0; i < 10; i++) {
    int c = i * 32 + lane;
    dst[c] = (half_t)((v[i] - mu) * rinv * g[c] + bta[c]);
  }
}

// ---------------- QKV GEMM: xw(131072x320) @ W^T(320x960) + b ------------
// writes padded q(n,d) k(n,d) vT(d,n) f16 tiles, d padded 40->64 (pre-zeroed)
__global__ __launch_bounds__(256) void k_gemm_qkv(const half_t* __restrict__ xw,
                                                  const half_t* __restrict__ wq,
                                                  const float* __restrict__ bias,
                                                  half_t* __restrict__ qkvp) {
  int wid = blockIdx.x * 8 + (threadIdx.x >> 5);
  int mt = wid / 15, ot = wid - mt * 15;
  int m0 = mt * 16, o0 = ot * 64;
  int lane = threadIdx.x & 31, r = lane & 15, hf = lane >> 4;

  v8f acc[4];
#pragma unroll
  for (int jn = 0; jn < 4; jn++)
#pragma unroll
    for (int j = 0; j < 8; j++) acc[jn][j] = 0.f;

  const half_t* abase = xw + (size_t)m0 * C_;
  for (int k0 = 0; k0 < C_; k0 += 32) {
    v16h a = load_a16(abase, C_, k0, r, hf);
#pragma unroll
    for (int jn = 0; jn < 4; jn++) {
      v16h bf = load_b16(wq + (size_t)(o0 + 16 * jn) * C_, C_, k0, r, hf);
      acc[jn] = wmma16(a, bf, acc[jn]);
    }
  }
  const float scale = 0.15811388300841897f;  // 40^-0.5
#pragma unroll
  for (int jn = 0; jn < 4; jn++) {
    int o = o0 + 16 * jn + r;
    int s = o / 320;
    int rem = o - s * 320;
    int h = rem / HD_;
    int d = rem - h * HD_;
    float bv = bias[o];
    float mul = (s == 0) ? scale : 1.0f;
#pragma unroll
    for (int j = 0; j < 8; j++) {
      int m = m0 + j + 8 * hf;
      int w = m >> 6, n = m & 63;
      float val = (acc[jn][j] + bv) * mul;
      size_t idx;
      if (s == 2)
        idx = ((size_t)((w * 3 + 2) * NH_ + h)) * 4096 + (size_t)d * 64 + n;
      else
        idx = ((size_t)((w * 3 + s) * NH_ + h)) * 4096 + (size_t)n * 64 + d;
      qkvp[idx] = (half_t)val;
    }
  }
}

// ---------------- windowed attention: one block per (window, head) -------
__global__ __launch_bounds__(128) void k_attn(const half_t* __restrict__ qkvp,
                                              const float* __restrict__ relb,
                                              half_t* __restrict__ ao) {
  __shared__ __align__(16) half_t pbuf[4][16][72];
  int w = blockIdx.x >> 3, h = blockIdx.x & 7;
  int t = threadIdx.x >> 5, lane = threadIdx.x & 31;
  int r = lane & 15, hf = lane >> 4;
  const half_t* Qb = qkvp + ((size_t)((w * 3 + 0) * NH_ + h)) * 4096;
  const half_t* Kb = qkvp + ((size_t)((w * 3 + 1) * NH_ + h)) * 4096;
  const half_t* Vt = qkvp + ((size_t)((w * 3 + 2) * NH_ + h)) * 4096;

  v16h q0 = load_a16(Qb + 16 * t * 64, 64, 0, r, hf);
  v16h q1 = load_a16(Qb + 16 * t * 64, 64, 32, r, hf);

  int win = w & 1023, wy = win >> 5, wx = win & 31;
  v8f acc[4];
#pragma unroll
  for (int jn = 0; jn < 4; jn++) {
    int col = 16 * jn + r;
    int jy = col >> 3, jx = col & 7;
    int cy = wy * 8 + jy, cx = wx * 8 + jx;
    int lcol = ((cy < 248) ? 0 : ((cy < 252) ? 1 : 2)) * 3 +
               ((cx < 248) ? 0 : ((cx < 252) ? 1 : 2));
#pragma unroll
    for (int j = 0; j < 8; j++) {
      int rowp = 16 * t + j + 8 * hf;
      int iy = rowp >> 3, ix = rowp & 7;
      int ry = wy * 8 + iy, rx = wx * 8 + ix;
      int lrow = ((ry < 248) ? 0 : ((ry < 252) ? 1 : 2)) * 3 +
                 ((rx < 248) ? 0 : ((rx < 252) ? 1 : 2));
      int rp = (iy - jy + 7) * 15 + (ix - jx + 7);
      acc[jn][j] = relb[rp * NH_ + h] + ((lrow != lcol) ? -100.0f : 0.0f);
    }
  }
#pragma unroll
  for (int jn = 0; jn < 4; jn++) {
    v16h kf0 = load_b16(Kb + 16 * jn * 64, 64, 0, r, hf);
    v16h kf1 = load_b16(Kb + 16 * jn * 64, 64, 32, r, hf);
    acc[jn] = wmma16(q1, kf1, acc[jn]);
    acc[jn] = wmma16(q0, kf0, acc[jn]);
  }
  // softmax: row of a D-tile lives across a 16-lane half
  float inv[8];
#pragma unroll
  for (int j = 0; j < 8; j++) {
    float m = fmaxf(fmaxf(acc[0][j], acc[1][j]), fmaxf(acc[2][j], acc[3][j]));
#pragma unroll
    for (int d = 8; d >= 1; d >>= 1) m = fmaxf(m, __shfl_xor(m, d, 32));
    float sum = 0.f;
#pragma unroll
    for (int jn = 0; jn < 4; jn++) {
      acc[jn][j] = __expf(acc[jn][j] - m);
      sum += acc[jn][j];
    }
#pragma unroll
    for (int d = 8; d >= 1; d >>= 1) sum += __shfl_xor(sum, d, 32);
    inv[j] = 1.0f / sum;
  }
#pragma unroll
  for (int jn = 0; jn < 4; jn++)
#pragma unroll
    for (int j = 0; j < 8; j++)
      pbuf[t][j + 8 * hf][16 * jn + r] = (half_t)(acc[jn][j] * inv[j]);
  __syncthreads();

  v16h p0, p1;
  ((vf4*)&p0)[0] = *(const vf4*)&pbuf[t][r][hf * 8];
  ((vf4*)&p0)[1] = *(const vf4*)&pbuf[t][r][hf * 8 + 16];
  ((vf4*)&p1)[0] = *(const vf4*)&pbuf[t][r][32 + hf * 8];
  ((vf4*)&p1)[1] = *(const vf4*)&pbuf[t][r][32 + hf * 8 + 16];

#pragma unroll
  for (int jd = 0; jd < 3; jd++) {
    v16h v0 = load_b16(Vt + 16 * jd * 64, 64, 0, r, hf);
    v16h v1 = load_b16(Vt + 16 * jd * 64, 64, 32, r, hf);
    v8f o;
#pragma unroll
    for (int j = 0; j < 8; j++) o[j] = 0.f;
    o = wmma16(p1, v1, o);
    o = wmma16(p0, v0, o);
    int d = 16 * jd + r;
    if (d < HD_) {
#pragma unroll
      for (int j = 0; j < 8; j++) {
        int rowg = 16 * t + j + 8 * hf;
        ao[(size_t)(w * 64 + rowg) * C_ + h * HD_ + d] = (half_t)o[j];
      }
    }
  }
}

// --------- proj GEMM + window reverse + roll(+4,+4) + shortcut -> d_out ---
__global__ __launch_bounds__(256) void k_gemm_proj(const half_t* __restrict__ ao,
                                                   const half_t* __restrict__ wp,
                                                   const float* __restrict__ pb,
                                                   const float* __restrict__ xin,
                                                   float* __restrict__ xout) {
  int wid = blockIdx.x * 8 + (threadIdx.x >> 5);
  int mt = wid / 5, ot = wid - mt * 5;
  int m0 = mt * 16, o0 = ot * 64;
  int lane = threadIdx.x & 31, r = lane & 15, hf = lane >> 4;
  v8f acc[4];
#pragma unroll
  for (int jn = 0; jn < 4; jn++)
#pragma unroll
    for (int j = 0; j < 8; j++) acc[jn][j] = 0.f;
  const half_t* abase = ao + (size_t)m0 * C_;
  for (int k0 = 0; k0 < C_; k0 += 32) {
    v16h a = load_a16(abase, C_, k0, r, hf);
#pragma unroll
    for (int jn = 0; jn < 4; jn++) {
      v16h bf = load_b16(wp + (size_t)(o0 + 16 * jn) * C_, C_, k0, r, hf);
      acc[jn] = wmma16(a, bf, acc[jn]);
    }
  }
#pragma unroll
  for (int jn = 0; jn < 4; jn++) {
    int c = o0 + 16 * jn + r;
    float bv = pb[c];
#pragma unroll
    for (int j = 0; j < 8; j++) {
      int m = m0 + j + 8 * hf;
      int w = m >> 6, n = m & 63;
      int b = w >> 10, win = w & 1023;
      int y = (((win >> 5) << 3) + (n >> 3) + SH_) & 255;
      int x = (((win & 31) << 3) + (n & 7) + SH_) & 255;
      size_t l = (size_t)b * 65536 + (size_t)y * 256 + x;
      size_t idx = l * C_ + c;
      xout[idx] = acc[jn][j] + bv + xin[idx];
    }
  }
}

// ---------------- LN2 (plain) -> yn f32 ----------------
__global__ __launch_bounds__(256) void k_ln2(const float* __restrict__ x2,
                                             const float* __restrict__ g,
                                             const float* __restrict__ bta,
                                             float* __restrict__ yn) {
  int wave = blockIdx.x * 8 + (threadIdx.x >> 5);
  int lane = threadIdx.x & 31;
  const float* row = x2 + (size_t)wave * C_;
  float v[10];
  float s = 0.f;
#pragma unroll
  for (int i = 0; i < 10; i++) { v[i] = row[i * 32 + lane]; s += v[i]; }
#pragma unroll
  for (int d = 16; d >= 1; d >>= 1) s += __shfl_xor(s, d, 32);
  float mu = s * (1.0f / C_);
  float vs = 0.f;
#pragma unroll
  for (int i = 0; i < 10; i++) { float t = v[i] - mu; vs += t * t; }
#pragma unroll
  for (int d = 16; d >= 1; d >>= 1) vs += __shfl_xor(vs, d, 32);
  float rinv = rsqrtf(vs * (1.0f / C_) + 1e-5f);
  float* dst = yn + (size_t)wave * C_;
#pragma unroll
  for (int i = 0; i < 10; i++) {
    int c = i * 32 + lane;
    dst[c] = (v[i] - mu) * rinv * g[c] + bta[c];
  }
}

// -------- depthwise 3x3x3 conv over (t,y,x) + residual in-place on d_out --
__global__ __launch_bounds__(256) void k_dwconv(const float* __restrict__ yn,
                                                const float* __restrict__ wgt,
                                                const float* __restrict__ cb,
                                                float* __restrict__ out) {
  size_t idx = (size_t)blockIdx.x * 256 + threadIdx.x;
  int c = (int)(idx % C_);
  size_t pl = idx / C_;
  int x = (int)(pl & 255);
  int y = (int)((pl >> 8) & 255);
  int b = (int)(pl >> 16);
  int g = c / 5, tt = c - g * 5;
  float sum = cb[g];
  const float* wg = wgt + g * 27;
#pragma unroll
  for (int dt = -1; dt <= 1; dt++) {
    int t2 = tt + dt;
    if (t2 < 0 || t2 >= 5) continue;
    int c2 = g * 5 + t2;
#pragma unroll
    for (int dy = -1; dy <= 1; dy++) {
      int y2 = y + dy;
      if ((unsigned)y2 >= 256u) continue;
#pragma unroll
      for (int dx = -1; dx <= 1; dx++) {
        int xx2 = x + dx;
        if ((unsigned)xx2 >= 256u) continue;
        float vv = yn[(((size_t)b * 65536) + (size_t)y2 * 256 + xx2) * C_ + c2];
        sum += wg[(dt + 1) * 9 + (dy + 1) * 3 + (dx + 1)] * vv;
      }
    }
  }
  out[idx] += sum;
}

// ---------------- host launcher ----------------
extern "C" void kernel_launch(void* const* d_in, const int* in_sizes, int n_in,
                              void* d_out, int out_size, void* d_ws,
                              size_t ws_size, hipStream_t stream) {
  (void)in_sizes; (void)n_in; (void)out_size; (void)ws_size;
  const float* x     = (const float*)d_in[0];
  const float* n1g   = (const float*)d_in[1];
  const float* n1b   = (const float*)d_in[2];
  const float* qkv_w = (const float*)d_in[3];
  const float* qkv_b = (const float*)d_in[4];
  const float* relb  = (const float*)d_in[5];
  const float* prj_w = (const float*)d_in[6];
  const float* prj_b = (const float*)d_in[7];
  const float* n2g   = (const float*)d_in[8];
  const float* n2b   = (const float*)d_in[9];
  const float* dw_w  = (const float*)d_in[10];
  const float* dw_b  = (const float*)d_in[11];
  float* out = (float*)d_out;  // holds x2 after proj, final result after conv

  char* ws = (char*)d_ws;
  const size_t QKVP_BYTES = 402653184ull;  // 2048*3*8*64*64 * 2B (padded qkv)
  const size_t XW_BYTES   = 83886080ull;   // 131072*320 * 2B
  half_t* qkvp = (half_t*)ws;                          // phase 2-3
  float*  yn   = (float*)ws;                            // phase 5-6 (reuse)
  half_t* xw   = (half_t*)(ws + QKVP_BYTES);            // phase 1-2
  half_t* ao   = xw;                                    // phase 3-4 (reuse)
  half_t* wqh  = (half_t*)(ws + QKVP_BYTES + XW_BYTES);
  half_t* wph  = wqh + 960 * 320;

  // zero the padded qkv buffer so head-dim 40->64 pad contributes 0 to WMMA K
  hipMemsetAsync(qkvp, 0, QKVP_BYTES, stream);

  k_cvt_h<<<(960 * 320 + 255) / 256, 256, 0, stream>>>(qkv_w, wqh, 960 * 320);
  k_cvt_h<<<(320 * 320 + 255) / 256, 256, 0, stream>>>(prj_w, wph, 320 * 320);

  k_ln1<<<16384, 256, 0, stream>>>(x, n1g, n1b, xw);                 // 131072 px
  k_gemm_qkv<<<15360, 256, 0, stream>>>(xw, wqh, qkv_b, qkvp);       // 8192x15 waves
  k_attn<<<16384, 128, 0, stream>>>(qkvp, relb, ao);                 // 2048x8
  k_gemm_proj<<<5120, 256, 0, stream>>>(ao, wph, prj_b, x, out);     // 8192x5 waves
  k_ln2<<<16384, 256, 0, stream>>>(out, n2g, n2b, yn);
  k_dwconv<<<163840, 256, 0, stream>>>(yn, dw_w, dw_b, out);         // 41.9M elems
}
